// StructureGNNLayer_36163624632830
// MI455X (gfx1250) — matrix operand order repaired and checked by reading, not compiled, not run on hardware
//
#include <hip/hip_runtime.h>

// ---------------------------------------------------------------------------
// StructureGNNLayer for MI455X (gfx1250, wave32, WMMA bf16 -> f32 accumulate)
// M-tile = 32 rows per block (2 x 16-row WMMA tiles) so every B fragment
// fetched from L2 feeds two v_wmma issues -> halves L2 weight traffic.
// ---------------------------------------------------------------------------

#define HID     256
#define EDIM    16
#define K1      528          // 2*HID + EDIM
#define K1P     544          // padded to multiple of 32
#define K2      512          // 2*HID
#define TPB     128          // 4 waves per block
#define MT      2            // M-tiles per block (32 rows)

typedef __attribute__((ext_vector_type(16))) __bf16 v16bf;
typedef __attribute__((ext_vector_type(8)))  __bf16 v8bf;
typedef __attribute__((ext_vector_type(8)))  float  v8f;

// f32 -> bf16 round-to-nearest-even (bit-level, no target-dependent cvt)
static __device__ __forceinline__ __bf16 f2bf(float f) {
  union { float f; unsigned u; } v; v.f = f;
  unsigned r = v.u + 0x7FFFu + ((v.u >> 16) & 1u);
  unsigned short s = (unsigned short)(r >> 16);
  return __builtin_bit_cast(__bf16, s);
}

static __device__ __forceinline__ float silu(float x) {
  return x / (1.0f + __expf(-x));
}

// A-fragment (16x32 bf16 tile in LDS), contiguous k order (same as B):
// lane = half*16 + m; elems 0..7 -> K = kb + half*16 + j,
//                     elems 8..15 -> K = kb + half*16 + 8 + j.
static __device__ __forceinline__ v16bf load_a_frag(const __bf16* row, int kb, int half) {
  v8bf lo = *reinterpret_cast<const v8bf*>(row + kb + half * 16);
  v8bf hi = *reinterpret_cast<const v8bf*>(row + kb + half * 16 + 8);
  v16bf a;
#pragma unroll
  for (int i = 0; i < 8; ++i) { a[i] = lo[i]; a[i + 8] = hi[i]; }
  return a;
}

// ---------------------------------------------------------------------------
// Prep kernels: weight transpose+convert, activation convert
// ---------------------------------------------------------------------------

// src: (K x N) row-major f32  ->  dst: (N x Kpad) column-major bf16 (zero pad)
__global__ void gnn_tcvt(const float* __restrict__ src, __bf16* __restrict__ dst,
                         int K, int Kpad, int N) {
  int i = blockIdx.x * 256 + threadIdx.x;
  if (i >= N * Kpad) return;
  int n = i / Kpad, k = i - n * Kpad;
  dst[(size_t)n * Kpad + k] = (k < K) ? f2bf(src[(size_t)k * N + n]) : f2bf(0.0f);
}

__global__ void gnn_cvt(const float* __restrict__ src, __bf16* __restrict__ dst,
                        unsigned int n) {
  unsigned int i = blockIdx.x * 256 + threadIdx.x;
  if (i < n) dst[i] = f2bf(src[i]);
}

// ---------------------------------------------------------------------------
// Edge message kernel: 32 edges / block, 4 waves, 2 chained WMMA GEMMs,
// atomic scatter-add into aggr. Messages never hit HBM as a dense tensor.
// ---------------------------------------------------------------------------
__global__ void __launch_bounds__(TPB)
gnn_msg(const __bf16* __restrict__ hbf,  const __bf16* __restrict__ eabf,
        const int* __restrict__ srcIdx,  const int* __restrict__ dstIdx,
        const __bf16* __restrict__ W1T,  const float* __restrict__ b1,
        const __bf16* __restrict__ W2T,  const float* __restrict__ b2,
        float* __restrict__ aggr) {
  __shared__ __align__(32) __bf16 sA[32 * K1P];     // 34816 B gathered input tile
  __shared__ __align__(32) __bf16 sH[32 * 272];     // 17408 B hidden tile (pad stride)
  __shared__ int sSrc[32], sDst[32];

  const int tid   = threadIdx.x;
  const int ebase = blockIdx.x * 32;

  if (tid < 32) {
    sSrc[tid] = srcIdx[ebase + tid];
    sDst[tid] = dstIdx[ebase + tid];
  }
  __syncthreads();

  // Stage A tile: rows = edges; [ h[src] (256) | h[dst] (256) | attr (16) | pad ]
  for (int i = tid; i < 32 * (K1P / 4); i += TPB) {
    int r  = i / (K1P / 4);
    int o4 = (i - r * (K1P / 4)) * 4;
    unsigned long long v = 0ull;
    if (o4 < 256)       v = *reinterpret_cast<const unsigned long long*>(hbf  + (size_t)sSrc[r] * HID + o4);
    else if (o4 < 512)  v = *reinterpret_cast<const unsigned long long*>(hbf  + (size_t)sDst[r] * HID + (o4 - 256));
    else if (o4 < K1)   v = *reinterpret_cast<const unsigned long long*>(eabf + (size_t)(ebase + r) * EDIM + (o4 - 512));
    *reinterpret_cast<unsigned long long*>(sA + r * K1P + o4) = v;
  }
  __syncthreads();

  const int wave = tid >> 5;
  const int lane = tid & 31;
  const int half = lane >> 4;
  const int nloc = lane & 15;

  // ---- GEMM 1: (32 x 544) @ (544 x 256); wave owns 64 cols x 2 M-tiles ----
  v8f acc[MT][4] = {};
  for (int kc = 0; kc < K1P / 32; ++kc) {
    int kb = kc * 32;
    v16bf a0 = load_a_frag(sA + nloc * K1P,        kb, half);
    v16bf a1 = load_a_frag(sA + (16 + nloc) * K1P, kb, half);
#pragma unroll
    for (int t = 0; t < 4; ++t) {
      int col = wave * 64 + t * 16 + nloc;
      v16bf b = *reinterpret_cast<const v16bf*>(W1T + (size_t)col * K1P + kb + half * 16);
      acc[0][t] = __builtin_amdgcn_wmma_f32_16x16x32_bf16(false, a0, false, b, (short)0, acc[0][t], false, false);
      acc[1][t] = __builtin_amdgcn_wmma_f32_16x16x32_bf16(false, a1, false, b, (short)0, acc[1][t], false, false);
    }
  }

  // bias + SiLU -> bf16 hidden tile in LDS
#pragma unroll
  for (int mt = 0; mt < MT; ++mt) {
#pragma unroll
    for (int t = 0; t < 4; ++t) {
      int col = wave * 64 + t * 16 + nloc;
      float bias = b1[col];
#pragma unroll
      for (int r = 0; r < 8; ++r) {
        int m = mt * 16 + r + half * 8;
        sH[m * 272 + col] = f2bf(silu(acc[mt][t][r] + bias));
      }
    }
  }
  __syncthreads();

  // ---- GEMM 2: (32 x 256) @ (256 x 256) ----
  v8f dcc[MT][4] = {};
  for (int kc = 0; kc < HID / 32; ++kc) {
    int kb = kc * 32;
    v16bf a0 = load_a_frag(sH + nloc * 272,        kb, half);
    v16bf a1 = load_a_frag(sH + (16 + nloc) * 272, kb, half);
#pragma unroll
    for (int t = 0; t < 4; ++t) {
      int col = wave * 64 + t * 16 + nloc;
      v16bf b = *reinterpret_cast<const v16bf*>(W2T + (size_t)col * HID + kb + half * 16);
      dcc[0][t] = __builtin_amdgcn_wmma_f32_16x16x32_bf16(false, a0, false, b, (short)0, dcc[0][t], false, false);
      dcc[1][t] = __builtin_amdgcn_wmma_f32_16x16x32_bf16(false, a1, false, b, (short)0, dcc[1][t], false, false);
    }
  }

  // bias + scatter-add by destination node (f32 atomics)
#pragma unroll
  for (int mt = 0; mt < MT; ++mt) {
#pragma unroll
    for (int t = 0; t < 4; ++t) {
      int col = wave * 64 + t * 16 + nloc;
      float bias = b2[col];
#pragma unroll
      for (int r = 0; r < 8; ++r) {
        int m = mt * 16 + r + half * 8;
        atomicAdd(&aggr[(size_t)sDst[m] * HID + col], dcc[mt][t][r] + bias);
      }
    }
  }
}

// ---------------------------------------------------------------------------
// Node update kernel: 32 nodes / block. [h | aggr] -> MLP -> residual ->
// LayerNorm -> mask, fully fused.
// ---------------------------------------------------------------------------
__global__ void __launch_bounds__(TPB)
gnn_upd(const __bf16* __restrict__ hbf, const float* __restrict__ hflat,
        const float* __restrict__ aggr,
        const __bf16* __restrict__ U1T, const float* __restrict__ c1,
        const __bf16* __restrict__ U2T, const float* __restrict__ c2,
        const float* __restrict__ gamma, const float* __restrict__ beta,
        const float* __restrict__ mask, float* __restrict__ out) {
  __shared__ __align__(32) __bf16 sA[32 * 528];   // K=512, padded stride
  __shared__ __align__(32) __bf16 sH[32 * 272];
  __shared__ float sO[32 * HID];
  __shared__ float sMu[32], sRs[32];

  const int tid = threadIdx.x;
  const int nb  = blockIdx.x * 32;

  // Stage A tile: [ h[node] bf16 (256) | bf16(aggr[node]) (256) ]
  for (int i = tid; i < 32 * (K2 / 4); i += TPB) {
    int r  = i / (K2 / 4);
    int o4 = (i - r * (K2 / 4)) * 4;
    int node = nb + r;
    if (o4 < 256) {
      *reinterpret_cast<unsigned long long*>(sA + r * 528 + o4) =
          *reinterpret_cast<const unsigned long long*>(hbf + (size_t)node * HID + o4);
    } else {
      float4 f = *reinterpret_cast<const float4*>(aggr + (size_t)node * HID + (o4 - 256));
      __bf16* p = sA + r * 528 + o4;
      p[0] = f2bf(f.x); p[1] = f2bf(f.y); p[2] = f2bf(f.z); p[3] = f2bf(f.w);
    }
  }
  __syncthreads();

  const int wave = tid >> 5;
  const int lane = tid & 31;
  const int half = lane >> 4;
  const int nloc = lane & 15;

  // ---- GEMM 1: (32 x 512) @ (512 x 256) ----
  v8f acc[MT][4] = {};
  for (int kc = 0; kc < K2 / 32; ++kc) {
    int kb = kc * 32;
    v16bf a0 = load_a_frag(sA + nloc * 528,        kb, half);
    v16bf a1 = load_a_frag(sA + (16 + nloc) * 528, kb, half);
#pragma unroll
    for (int t = 0; t < 4; ++t) {
      int col = wave * 64 + t * 16 + nloc;
      v16bf b = *reinterpret_cast<const v16bf*>(U1T + (size_t)col * K2 + kb + half * 16);
      acc[0][t] = __builtin_amdgcn_wmma_f32_16x16x32_bf16(false, a0, false, b, (short)0, acc[0][t], false, false);
      acc[1][t] = __builtin_amdgcn_wmma_f32_16x16x32_bf16(false, a1, false, b, (short)0, acc[1][t], false, false);
    }
  }
#pragma unroll
  for (int mt = 0; mt < MT; ++mt) {
#pragma unroll
    for (int t = 0; t < 4; ++t) {
      int col = wave * 64 + t * 16 + nloc;
      float bias = c1[col];
#pragma unroll
      for (int r = 0; r < 8; ++r) {
        int m = mt * 16 + r + half * 8;
        sH[m * 272 + col] = f2bf(silu(acc[mt][t][r] + bias));
      }
    }
  }
  __syncthreads();

  // ---- GEMM 2: (32 x 256) @ (256 x 256) + residual ----
  v8f dcc[MT][4] = {};
  for (int kc = 0; kc < HID / 32; ++kc) {
    int kb = kc * 32;
    v16bf a0 = load_a_frag(sH + nloc * 272,        kb, half);
    v16bf a1 = load_a_frag(sH + (16 + nloc) * 272, kb, half);
#pragma unroll
    for (int t = 0; t < 4; ++t) {
      int col = wave * 64 + t * 16 + nloc;
      v16bf b = *reinterpret_cast<const v16bf*>(U2T + (size_t)col * HID + kb + half * 16);
      dcc[0][t] = __builtin_amdgcn_wmma_f32_16x16x32_bf16(false, a0, false, b, (short)0, dcc[0][t], false, false);
      dcc[1][t] = __builtin_amdgcn_wmma_f32_16x16x32_bf16(false, a1, false, b, (short)0, dcc[1][t], false, false);
    }
  }
#pragma unroll
  for (int mt = 0; mt < MT; ++mt) {
#pragma unroll
    for (int t = 0; t < 4; ++t) {
      int col = wave * 64 + t * 16 + nloc;
      float bias = c2[col];
#pragma unroll
      for (int r = 0; r < 8; ++r) {
        int m = mt * 16 + r + half * 8;
        sO[m * HID + col] = hflat[(size_t)(nb + m) * HID + col] + dcc[mt][t][r] + bias;
      }
    }
  }
  __syncthreads();

  // LayerNorm statistics: one thread per row (32 active = 1 wave)
  if (tid < 32) {
    const float* row = sO + tid * HID;
    float s = 0.0f;
    for (int j = 0; j < HID; ++j) s += row[j];
    float mu = s * (1.0f / HID);
    float v = 0.0f;
    for (int j = 0; j < HID; ++j) { float d = row[j] - mu; v += d * d; }
    sMu[tid] = mu;
    sRs[tid] = __frsqrt_rn(v * (1.0f / HID) + 1e-5f);
  }
  __syncthreads();

  // normalize, affine, mask, write out
  for (int i = tid; i < 32 * HID; i += TPB) {
    int r = i >> 8, col = i & (HID - 1);
    int node = nb + r;
    float y = (sO[i] - sMu[r]) * sRs[r] * gamma[col] + beta[col];
    out[(size_t)node * HID + col] = y * mask[node];
  }
}

// ---------------------------------------------------------------------------
// Launch
// ---------------------------------------------------------------------------
extern "C" void kernel_launch(void* const* d_in, const int* in_sizes, int n_in,
                              void* d_out, int out_size, void* d_ws, size_t ws_size,
                              hipStream_t stream) {
  const float* h     = (const float*)d_in[0];
  const int*   eidx  = (const int*)d_in[1];      // (2, E): row0 = src, row1 = dst
  const float* ea    = (const float*)d_in[2];
  const float* mask  = (const float*)d_in[3];
  const float* W1    = (const float*)d_in[4];
  const float* b1    = (const float*)d_in[5];
  const float* W2    = (const float*)d_in[6];
  const float* b2    = (const float*)d_in[7];
  const float* U1    = (const float*)d_in[8];
  const float* c1    = (const float*)d_in[9];
  const float* U2    = (const float*)d_in[10];
  const float* c2    = (const float*)d_in[11];
  const float* gamma = (const float*)d_in[12];
  const float* beta  = (const float*)d_in[13];

  const int E  = in_sizes[1] / 2;
  const int BN = in_sizes[0] / HID;
  const int* srcIdx = eidx;
  const int* dstIdx = eidx + E;

  // workspace carve-up (256B aligned)
  char* ws = (char*)d_ws;
  size_t off = 0;
  auto carve = [&](size_t bytes) {
    void* p = ws + off;
    off += (bytes + 255) & ~(size_t)255;
    return p;
  };
  __bf16* W1T  = (__bf16*)carve((size_t)HID * K1P * 2);
  __bf16* W2T  = (__bf16*)carve((size_t)HID * HID * 2);
  __bf16* U1T  = (__bf16*)carve((size_t)HID * K2  * 2);
  __bf16* U2T  = (__bf16*)carve((size_t)HID * HID * 2);
  __bf16* hbf  = (__bf16*)carve((size_t)BN * HID * 2);
  __bf16* eabf = (__bf16*)carve((size_t)E * EDIM * 2);
  float*  aggr = (float*)carve((size_t)BN * HID * 4);
  (void)ws_size; (void)n_in; (void)out_size;

  // 1) weights -> bf16, transposed (column-major), K padded where needed
  gnn_tcvt<<<(HID * K1P + 255) / 256, 256, 0, stream>>>(W1, W1T, K1, K1P, HID);
  gnn_tcvt<<<(HID * HID + 255) / 256, 256, 0, stream>>>(W2, W2T, HID, HID, HID);
  gnn_tcvt<<<(HID * K2  + 255) / 256, 256, 0, stream>>>(U1, U1T, K2, K2, HID);
  gnn_tcvt<<<(HID * HID + 255) / 256, 256, 0, stream>>>(U2, U2T, HID, HID, HID);

  // 2) activations -> bf16 copies (halves random-gather traffic; L2-resident)
  unsigned int nh = (unsigned int)BN * HID;
  unsigned int ne = (unsigned int)E * EDIM;
  gnn_cvt<<<(nh + 255) / 256, 256, 0, stream>>>(h, hbf, nh);
  gnn_cvt<<<(ne + 255) / 256, 256, 0, stream>>>(ea, eabf, ne);

  // 3) zero aggregation buffer
  hipMemsetAsync(aggr, 0, (size_t)BN * HID * 4, stream);

  // 4) edge messages + scatter-add (E/32 blocks)
  gnn_msg<<<E / 32, TPB, 0, stream>>>(hbf, eabf, srcIdx, dstIdx,
                                      W1T, b1, W2T, b2, aggr);

  // 5) node update + LayerNorm + mask (BN/32 blocks)
  gnn_upd<<<BN / 32, TPB, 0, stream>>>(hbf, h, aggr, U1T, c1, U2T, c2,
                                       gamma, beta, mask, (float*)d_out);
}
